// TaskMoE_80900003987999
// MI455X (gfx1250) — matrix operand order repaired and checked by reading, not compiled, not run on hardware
//
#include <hip/hip_runtime.h>
#include <hip/hip_bf16.h>
#include <math.h>

// ---------------------------------------------------------------------------
// Problem constants (from reference): B,H,W,D = 4,32,64,512 ; E=8 ; F=2048
// ---------------------------------------------------------------------------
#define N_TOK 8192
#define DDIM  512
#define FDIM  2048
#define NEXP  8
#define HALF_N3 12288u   // (N_TOK*3)/2, JAX threefry counter split point

typedef __attribute__((ext_vector_type(16))) _Float16 v16h;
typedef __attribute__((ext_vector_type(8)))  float    v8f;

// Workspace layout (bytes)
#define WS_COUNTS   0
#define WS_PMAX     4096                       // N floats   (32 KB)
#define WS_IDX      65536                      // E*N ints   (256 KB)
#define WS_WP1      (1u << 20)                 // 16 MB f16 packed W1
#define WS_WP2      ((1u << 20) + (16u << 20)) // 16 MB f16 packed W2
#define WS_NEEDED   ((size_t)(1u << 20) + (32u << 20))

// ---------------------------------------------------------------------------
// Threefry-2x32 (JAX default PRNG), key = jax.random.key(42) -> (0, 42)
// ---------------------------------------------------------------------------
#define TF_RND(x0,x1,r) { x0 += x1; x1 = ((x1<<(r))|(x1>>(32-(r)))); x1 ^= x0; }

__device__ __forceinline__ void threefry2x32(unsigned k0, unsigned k1,
                                             unsigned c0, unsigned c1,
                                             unsigned& o0, unsigned& o1) {
  unsigned ks2 = k0 ^ k1 ^ 0x1BD11BDAu;
  unsigned x0 = c0 + k0, x1 = c1 + k1;
  TF_RND(x0,x1,13) TF_RND(x0,x1,15) TF_RND(x0,x1,26) TF_RND(x0,x1,6)
  x0 += k1;  x1 += ks2 + 1u;
  TF_RND(x0,x1,17) TF_RND(x0,x1,29) TF_RND(x0,x1,16) TF_RND(x0,x1,24)
  x0 += ks2; x1 += k0 + 2u;
  TF_RND(x0,x1,13) TF_RND(x0,x1,15) TF_RND(x0,x1,26) TF_RND(x0,x1,6)
  x0 += k0;  x1 += k1 + 3u;
  TF_RND(x0,x1,17) TF_RND(x0,x1,29) TF_RND(x0,x1,16) TF_RND(x0,x1,24)
  x0 += k1;  x1 += ks2 + 4u;
  TF_RND(x0,x1,13) TF_RND(x0,x1,15) TF_RND(x0,x1,26) TF_RND(x0,x1,6)
  x0 += ks2; x1 += k0 + 5u;
  o0 = x0; o1 = x1;
}

__device__ __forceinline__ float jax_uniform_bits(unsigned i) {
  bool lo = i < HALF_N3;
  unsigned a0 = lo ? i : (i - HALF_N3);
  unsigned a1 = a0 + HALF_N3;
  unsigned o0, o1;
  threefry2x32(0u, 42u, a0, a1, o0, o1);
  unsigned bits = lo ? o0 : o1;
  unsigned fb = (bits >> 9) | 0x3f800000u;
  return __uint_as_float(fb) - 1.0f;
}

// ---------------------------------------------------------------------------
// Branchless exact-GELU: 0.5*x*(1+erf(x/sqrt2)), erf via A&S 7.1.26
// (|err| < 1.5e-7, below f16 rounding). Uses v_rcp_f32 + v_exp_f32, no exec
// divergence (unlike libm erff which emits s_and_saveexec polynomial splits).
// ---------------------------------------------------------------------------
__device__ __forceinline__ float gelu_exact(float x) {
  float z  = x * 0.70710678118654752440f;
  float az = fabsf(z);
  float t  = __builtin_amdgcn_rcpf(fmaf(0.3275911f, az, 1.0f));
  float p  = fmaf(t, 1.061405429f, -1.453152027f);
  p = fmaf(t, p,  1.421413741f);
  p = fmaf(t, p, -0.284496736f);
  p = fmaf(t, p,  0.254829592f);
  p = p * t;
  float e      = __expf(-z * z);
  float erfabs = fmaf(-p, e, 1.0f);
  float erfz   = copysignf(erfabs, z);
  return 0.5f * x * (1.0f + erfz);
}

// ---------------------------------------------------------------------------
// CDNA5 16x16x32 f16 fragment layouts (wave32):
//  A: lane row = lane&15; off = (lane&16)?8:0; value q -> K = off + q + (q>=8?8:0)
//  B: lane col = lane&15; K = off + q + (q>=8?8:0), same k order as A
//  C: vgpr r, lane<16 -> (M=r, N=lane); lane>=16 -> (M=8+r, N=lane-16)
// Packed tiles store each lane's 16 f16 values contiguously: [tile][lane][q].
// ---------------------------------------------------------------------------

// A-fragment-packed address for element (row m in 16-row set, K index Kf)
__device__ __forceinline__ int hpack_addr(int m, int Kf) {
  int kk = Kf & 31;
  int lp = m + ((kk & 8) ? 16 : 0);
  int qp = (kk & 16) ? (8 + (kk & 7)) : (kk & 7);
  return (Kf >> 5) * 512 + lp * 16 + qp;
}

// generic (unpacked) loaders for the fallback path ---------------------------
__device__ __forceinline__ v16h load_fragA_lds(const _Float16* p, int off) {
  v16h f;
#pragma unroll
  for (int j = 0; j < 4; ++j) {
    f[2*j]     = p[off + 2*j];
    f[2*j + 1] = p[off + 2*j + 1];
  }
#pragma unroll
  for (int j = 0; j < 4; ++j) {
    f[8 + 2*j] = p[16 + off + 2*j];
    f[9 + 2*j] = p[16 + off + 2*j + 1];
  }
  return f;
}

__device__ __forceinline__ v16h load_fragB_f32(const float* __restrict__ p, int ldn) {
  v16h f;
#pragma unroll
  for (int j = 0; j < 8; ++j) {
    f[2*j]     = (_Float16)p[(size_t)(2*j + ((j & 4) ? 8 : 0))     * ldn];
    f[2*j + 1] = (_Float16)p[(size_t)(2*j + 1 + ((j & 4) ? 8 : 0)) * ldn];
  }
  return f;
}

// ---------------------------------------------------------------------------
// Kernel 0: zero expert counters
// ---------------------------------------------------------------------------
__global__ void taskmoe_zero_counts(int* counts) {
  if (threadIdx.x < NEXP) counts[threadIdx.x] = 0;
}

// ---------------------------------------------------------------------------
// Kernel 1: router (jitter + logits + softmax-max + argmax + bucketing)
// ---------------------------------------------------------------------------
__global__ __launch_bounds__(256)
void taskmoe_router(const float* __restrict__ attr,
                    const float* __restrict__ Wr,
                    const float* __restrict__ br,
                    int* __restrict__ counts,
                    int* __restrict__ idxlist,
                    float* __restrict__ pmax) {
  int t = blockIdx.x * blockDim.x + threadIdx.x;
  if (t >= N_TOK) return;

  float na[3];
#pragma unroll
  for (int j = 0; j < 3; ++j) {
    float u = jax_uniform_bits(3u * (unsigned)t + (unsigned)j);
    na[j] = attr[t*3 + j] * (1.01f - 0.02f * u);
  }

  float lg[NEXP];
  float lmax = -1e30f;
  int   arg  = 0;
#pragma unroll
  for (int e = 0; e < NEXP; ++e) {
    float s = br[e];
#pragma unroll
    for (int j = 0; j < 3; ++j) s += na[j] * Wr[j*NEXP + e];
    lg[e] = s;
    if (s > lmax) { lmax = s; arg = e; }
  }
  float se = 0.0f;
#pragma unroll
  for (int e = 0; e < NEXP; ++e) se += expf(lg[e] - lmax);

  pmax[t] = 1.0f / se;
  int slot = atomicAdd(&counts[arg], 1);
  idxlist[arg * N_TOK + slot] = t;
}

// ---------------------------------------------------------------------------
// Kernels 2a/2b: pack W1/W2 fp32 -> f16 in B-fragment order.
//   Wp1[e][ct<F/16][kt<D/32][lane<32][q<16], elem = W1[e][kt*32+(lane>>4)*8+q+(q>=8?8:0)][ct*16+(lane&15)]
// ---------------------------------------------------------------------------
__global__ __launch_bounds__(256)
void taskmoe_pack_w1(const float* __restrict__ W1, _Float16* __restrict__ Wp1) {
  const size_t total = (size_t)NEXP * (FDIM/16) * (DDIM/32) * 512;
  for (size_t i = (size_t)blockIdx.x * 256 + threadIdx.x; i < total;
       i += (size_t)gridDim.x * 256) {
    unsigned q = (unsigned)(i & 15), lane = (unsigned)((i >> 4) & 31);
    size_t tile = i >> 9;
    unsigned kt = (unsigned)(tile & (DDIM/32 - 1));
    size_t t2 = tile >> 4;
    unsigned ct = (unsigned)(t2 & (FDIM/16 - 1));
    unsigned e  = (unsigned)(t2 >> 7);
    unsigned k = kt*32 + ((lane >> 4) << 3) + q + ((q & 8) ? 8 : 0);
    unsigned n = ct*16 + (lane & 15);
    Wp1[i] = (_Float16)W1[((size_t)e * DDIM + k) * FDIM + n];
  }
}

__global__ __launch_bounds__(256)
void taskmoe_pack_w2(const float* __restrict__ W2, _Float16* __restrict__ Wp2) {
  const size_t total = (size_t)NEXP * (DDIM/16) * (FDIM/32) * 512;
  for (size_t i = (size_t)blockIdx.x * 256 + threadIdx.x; i < total;
       i += (size_t)gridDim.x * 256) {
    unsigned q = (unsigned)(i & 15), lane = (unsigned)((i >> 4) & 31);
    size_t tile = i >> 9;
    unsigned kt = (unsigned)(tile & (FDIM/32 - 1));
    size_t t2 = tile >> 6;
    unsigned ct = (unsigned)(t2 & (DDIM/16 - 1));
    unsigned e  = (unsigned)(t2 >> 5);
    unsigned k = kt*32 + ((lane >> 4) << 3) + q + ((q & 8) ? 8 : 0);
    unsigned n = ct*16 + (lane & 15);
    Wp2[i] = (_Float16)W2[((size_t)e * FDIM + k) * DDIM + n];
  }
}

// ---------------------------------------------------------------------------
// Kernel 3 (fast path): expert FFN on 32-token tiles, 2x2 register blocking.
// 256 threads = 8 waves. Per k-step: 2 B frags + 2 A frags -> 4 WMMAs
// (1 global b128 + 1 ds b128 per wmma). LDS: X 32KB + H 128KB = 160KB.
// ---------------------------------------------------------------------------
__global__ __launch_bounds__(256)
void taskmoe_ffn_packed(const float* __restrict__ x,
                        const _Float16* __restrict__ Wp1,
                        const float* __restrict__ b1,
                        const _Float16* __restrict__ Wp2,
                        const float* __restrict__ b2,
                        const int*   __restrict__ counts,
                        const int*   __restrict__ idxlist,
                        const float* __restrict__ pmax,
                        float* __restrict__ out) {
  extern __shared__ __attribute__((aligned(32))) char smem[];
  _Float16* sAp = (_Float16*)smem;                      // [2][D/32][512]  32 KB
  _Float16* sHp = (_Float16*)(smem + 2*(DDIM/32)*512*2);// [2][F/32][512] 128 KB
  __shared__ int   sTok[32];
  __shared__ float sP[32];

  const int e    = blockIdx.x >> 8;        // 256 tiles of 32 tokens per expert
  const int tile = blockIdx.x & 255;
  const int cnt  = counts[e];
  if (tile * 32 >= cnt) return;            // uniform early-exit

  const int tid = threadIdx.x;

  if (tid < 32) {
    int s   = tile * 32 + tid;
    int tok = (s < cnt) ? idxlist[e * N_TOK + s] : -1;
    sTok[tid] = tok;
    sP[tid]   = (tok >= 0) ? pmax[tok] : 0.0f;
  }
  __syncthreads();

  // Stage gathered X (32 rows x 512) into A-fragment-packed LDS, f32 -> f16.
  for (int i = tid; i < 2 * (DDIM/32) * 512; i += 256) {
    int q    = i & 15;
    int lane = (i >> 4) & 31;
    int kt   = (i >> 9) & (DDIM/32 - 1);
    int s    = i >> 13;
    int off  = (lane & 16) ? 8 : 0;
    int K    = kt*32 + off + q + ((q & 8) ? 8 : 0);
    int row  = s*16 + (lane & 15);
    int tok  = sTok[row];
    float v  = (tok >= 0) ? x[(size_t)tok * DDIM + K] : 0.0f;
    sAp[i] = (_Float16)v;
  }
  __syncthreads();

  const int lane  = tid & 31;
  const int wave  = tid >> 5;
  const int nlo   = lane & 15;
  const int mbase = (lane & 16) ? 8 : 0;

  // ---------------- GEMM1: H = gelu(X @ W1[e] + b1[e]),  K = 512 -------------
  const _Float16* Wp1e = Wp1 + (size_t)e * (FDIM/16) * (DDIM/32) * 512;
  const float*    b1e  = b1  + (size_t)e * FDIM;

  for (int t = 0; t < 8; ++t) {              // 64 col-tile pairs, 8 per wave
    const int ct0 = (t * 8 + wave) * 2;
    const int ct1 = ct0 + 1;
    const float bias0 = b1e[ct0 * 16 + nlo];
    const float bias1 = b1e[ct1 * 16 + nlo];
    v8f acc00, acc01, acc10, acc11;
#pragma unroll
    for (int r = 0; r < 8; ++r) {
      acc00[r] = bias0; acc01[r] = bias1;
      acc10[r] = bias0; acc11[r] = bias1;
    }

    const _Float16* Bt0 = Wp1e + (size_t)ct0 * (DDIM/32) * 512 + lane * 16;
    const _Float16* Bt1 = Wp1e + (size_t)ct1 * (DDIM/32) * 512 + lane * 16;
#pragma unroll 2
    for (int kt = 0; kt < DDIM/32; ++kt) {
      v16h b0 = *(const v16h*)(Bt0 + (size_t)kt * 512);
      v16h b1f = *(const v16h*)(Bt1 + (size_t)kt * 512);
      v16h a0 = *(const v16h*)(sAp + (size_t)kt * 512 + lane * 16);
      v16h a1 = *(const v16h*)(sAp + (size_t)(DDIM/32 + kt) * 512 + lane * 16);
      acc00 = __builtin_amdgcn_wmma_f32_16x16x32_f16(false, a0, false, b0,  (short)0, acc00, false, false);
      acc01 = __builtin_amdgcn_wmma_f32_16x16x32_f16(false, a0, false, b1f, (short)0, acc01, false, false);
      acc10 = __builtin_amdgcn_wmma_f32_16x16x32_f16(false, a1, false, b0,  (short)0, acc10, false, false);
      acc11 = __builtin_amdgcn_wmma_f32_16x16x32_f16(false, a1, false, b1f, (short)0, acc11, false, false);
    }

    // branchless exact GELU, scatter into A-fragment-packed H for GEMM2.
#pragma unroll
    for (int r = 0; r < 8; ++r) {
      const int m   = mbase + r;
      const int Kf0 = ct0 * 16 + nlo;
      const int Kf1 = ct1 * 16 + nlo;
      const int ha0 = hpack_addr(m, Kf0);
      const int ha1 = hpack_addr(m, Kf1);
      const int SET1 = (FDIM/32) * 512;
      sHp[ha0]        = (_Float16)gelu_exact(acc00[r]);
      sHp[ha1]        = (_Float16)gelu_exact(acc01[r]);
      sHp[SET1 + ha0] = (_Float16)gelu_exact(acc10[r]);
      sHp[SET1 + ha1] = (_Float16)gelu_exact(acc11[r]);
    }
  }
  __syncthreads();

  // ---------------- GEMM2: Y = H @ W2[e] + b2[e],  K = 2048 ------------------
  const _Float16* Wp2e = Wp2 + (size_t)e * (DDIM/16) * (FDIM/32) * 512;
  const float*    b2e  = b2  + (size_t)e * DDIM;

  for (int t = 0; t < 2; ++t) {              // 16 col-tile pairs, 2 per wave
    const int ct0 = (t * 8 + wave) * 2;
    const int ct1 = ct0 + 1;
    const float bias0 = b2e[ct0 * 16 + nlo];
    const float bias1 = b2e[ct1 * 16 + nlo];
    v8f acc00, acc01, acc10, acc11;
#pragma unroll
    for (int r = 0; r < 8; ++r) {
      acc00[r] = bias0; acc01[r] = bias1;
      acc10[r] = bias0; acc11[r] = bias1;
    }

    const _Float16* Bt0 = Wp2e + (size_t)ct0 * (FDIM/32) * 512 + lane * 16;
    const _Float16* Bt1 = Wp2e + (size_t)ct1 * (FDIM/32) * 512 + lane * 16;
#pragma unroll 2
    for (int kt = 0; kt < FDIM/32; ++kt) {
      v16h b0 = *(const v16h*)(Bt0 + (size_t)kt * 512);
      v16h b1f = *(const v16h*)(Bt1 + (size_t)kt * 512);
      v16h a0 = *(const v16h*)(sHp + (size_t)kt * 512 + lane * 16);
      v16h a1 = *(const v16h*)(sHp + (size_t)(FDIM/32 + kt) * 512 + lane * 16);
      acc00 = __builtin_amdgcn_wmma_f32_16x16x32_f16(false, a0, false, b0,  (short)0, acc00, false, false);
      acc01 = __builtin_amdgcn_wmma_f32_16x16x32_f16(false, a0, false, b1f, (short)0, acc01, false, false);
      acc10 = __builtin_amdgcn_wmma_f32_16x16x32_f16(false, a1, false, b0,  (short)0, acc10, false, false);
      acc11 = __builtin_amdgcn_wmma_f32_16x16x32_f16(false, a1, false, b1f, (short)0, acc11, false, false);
    }

#pragma unroll
    for (int r = 0; r < 8; ++r) {
      const int m0 = mbase + r, m1 = 16 + mbase + r;
      const int tok0 = sTok[m0], tok1 = sTok[m1];
      const int col0 = ct0 * 16 + nlo, col1 = ct1 * 16 + nlo;
      if (tok0 >= 0) {
        out[(size_t)tok0 * DDIM + col0] = acc00[r] * sP[m0];
        out[(size_t)tok0 * DDIM + col1] = acc01[r] * sP[m0];
      }
      if (tok1 >= 0) {
        out[(size_t)tok1 * DDIM + col0] = acc10[r] * sP[m1];
        out[(size_t)tok1 * DDIM + col1] = acc11[r] * sP[m1];
      }
    }
  }
}

// ---------------------------------------------------------------------------
// Kernel 3 (fallback, small workspace): direct-from-fp32, 16-token tiles.
// ---------------------------------------------------------------------------
__global__ __launch_bounds__(256)
void taskmoe_ffn_direct(const float* __restrict__ x,
                        const float* __restrict__ W1,
                        const float* __restrict__ b1,
                        const float* __restrict__ W2,
                        const float* __restrict__ b2,
                        const int*   __restrict__ counts,
                        const int*   __restrict__ idxlist,
                        const float* __restrict__ pmax,
                        float* __restrict__ out) {
  extern __shared__ __attribute__((aligned(32))) char smem[];
  _Float16* sA = (_Float16*)smem;                       // [16][512]
  _Float16* sH = (_Float16*)(smem + 16*DDIM*2);         // [16][2048]
  __shared__ int   sTok[16];
  __shared__ float sP[16];

  const int e    = blockIdx.x >> 9;
  const int tile = blockIdx.x & 511;
  const int cnt  = counts[e];
  if (tile * 16 >= cnt) return;

  const int tid = threadIdx.x;
  if (tid < 16) {
    int s   = tile * 16 + tid;
    int tok = (s < cnt) ? idxlist[e * N_TOK + s] : -1;
    sTok[tid] = tok;
    sP[tid]   = (tok >= 0) ? pmax[tok] : 0.0f;
  }
  __syncthreads();

  for (int i = tid; i < 16 * DDIM; i += 256) {
    int r = i >> 9, c = i & (DDIM - 1);
    int tok = sTok[r];
    float v = (tok >= 0) ? x[(size_t)tok * DDIM + c] : 0.0f;
    sA[i] = (_Float16)v;
  }
  __syncthreads();

  const int lane  = tid & 31;
  const int wave  = tid >> 5;
  const int nlo   = lane & 15;
  const int koffA = (lane & 16) ? 8  : 0;
  const int koffB = (lane & 16) ? 8  : 0;
  const int mbase = (lane & 16) ? 8  : 0;

  const float* W1e = W1 + (size_t)e * DDIM * FDIM;
  const float* b1e = b1 + (size_t)e * FDIM;

  for (int t = 0; t < 16; ++t) {
    const int col = (t * 8 + wave) * 16;
    const float bias = b1e[col + nlo];
    v8f acc;
#pragma unroll
    for (int r = 0; r < 8; ++r) acc[r] = bias;
#pragma unroll 4
    for (int k = 0; k < DDIM; k += 32) {
      v16h a = load_fragA_lds(sA + (size_t)nlo * DDIM + k, koffA);
      v16h b = load_fragB_f32(W1e + (size_t)(k + koffB) * FDIM + col + nlo, FDIM);
      acc = __builtin_amdgcn_wmma_f32_16x16x32_f16(
          false, a, false, b, (short)0, acc, false, false);
    }
#pragma unroll
    for (int r = 0; r < 8; ++r) {
      sH[(size_t)(mbase + r) * FDIM + col + nlo] = (_Float16)gelu_exact(acc[r]);
    }
  }
  __syncthreads();

  const float* W2e = W2 + (size_t)e * FDIM * DDIM;
  const float* b2e = b2 + (size_t)e * DDIM;

  for (int t = 0; t < 4; ++t) {
    const int col = (t * 8 + wave) * 16;
    const float bias = b2e[col + nlo];
    v8f acc;
#pragma unroll
    for (int r = 0; r < 8; ++r) acc[r] = bias;
#pragma unroll 4
    for (int k = 0; k < FDIM; k += 32) {
      v16h a = load_fragA_lds(sH + (size_t)nlo * FDIM + k, koffA);
      v16h b = load_fragB_f32(W2e + (size_t)(k + koffB) * DDIM + col + nlo, DDIM);
      acc = __builtin_amdgcn_wmma_f32_16x16x32_f16(
          false, a, false, b, (short)0, acc, false, false);
    }
#pragma unroll
    for (int r = 0; r < 8; ++r) {
      int m   = mbase + r;
      int tok = sTok[m];
      if (tok >= 0) out[(size_t)tok * DDIM + col + nlo] = acc[r] * sP[m];
    }
  }
}

// ---------------------------------------------------------------------------
// Launch
// ---------------------------------------------------------------------------
extern "C" void kernel_launch(void* const* d_in, const int* in_sizes, int n_in,
                              void* d_out, int out_size, void* d_ws, size_t ws_size,
                              hipStream_t stream) {
  const float* x    = (const float*)d_in[0];
  const float* attr = (const float*)d_in[1];
  const float* Wr   = (const float*)d_in[2];
  const float* br   = (const float*)d_in[3];
  const float* W1   = (const float*)d_in[4];
  const float* b1   = (const float*)d_in[5];
  const float* W2   = (const float*)d_in[6];
  const float* b2   = (const float*)d_in[7];
  float* out = (float*)d_out;

  char* ws = (char*)d_ws;
  int*   counts  = (int*)(ws + WS_COUNTS);
  float* pmax    = (float*)(ws + WS_PMAX);
  int*   idxlist = (int*)(ws + WS_IDX);

  taskmoe_zero_counts<<<1, 32, 0, stream>>>(counts);
  taskmoe_router<<<N_TOK / 256, 256, 0, stream>>>(attr, Wr, br, counts, idxlist, pmax);

  if (ws_size >= WS_NEEDED) {
    _Float16* Wp1 = (_Float16*)(ws + WS_WP1);
    _Float16* Wp2 = (_Float16*)(ws + WS_WP2);
    taskmoe_pack_w1<<<4096, 256, 0, stream>>>(W1, Wp1);
    taskmoe_pack_w2<<<4096, 256, 0, stream>>>(W2, Wp2);
    const size_t smem = (size_t)2 * (DDIM/32) * 512 * 2 +
                        (size_t)2 * (FDIM/32) * 512 * 2;     // 160 KB
    taskmoe_ffn_packed<<<NEXP * (N_TOK / 32), 256, smem, stream>>>(
        x, Wp1, b1, Wp2, b2, counts, idxlist, pmax, out);
  } else {
    const size_t smem = (size_t)16 * DDIM * 2 + (size_t)16 * FDIM * 2;  // 80 KB
    taskmoe_ffn_direct<<<NEXP * (N_TOK / 16), 256, smem, stream>>>(
        x, W1, b1, W2, b2, counts, idxlist, pmax, out);
  }
}
